// GraphGenerator_1872605741593
// MI455X (gfx1250) — compile-verified, matrix operand order checked by reading, not run on hardware
//
#include <hip/hip_runtime.h>
#include <hip/hip_bf16.h>
#include <hip/hip_fp16.h>

typedef __attribute__((ext_vector_type(16))) _Float16 v16h;
typedef __attribute__((ext_vector_type(8)))  float    v8f;

#define LDH 32   // fixed row stride (floats) for all hidden-state buffers

// ---------------------------------------------------------------------------
// Utility: zero a float buffer
// ---------------------------------------------------------------------------
__global__ void k_zero_f32(float* __restrict__ p, int n) {
    int i = blockIdx.x * blockDim.x + threadIdx.x;
    if (i < n) p[i] = 0.0f;
}

// Pack W[K,F] into zero-padded Wp[Kpad,Fpad]
__global__ void k_pack_w(const float* __restrict__ W, float* __restrict__ Wp,
                         int K, int F, int Kpad, int Fpad) {
    int i = blockIdx.x * blockDim.x + threadIdx.x;
    if (i >= Kpad * Fpad) return;
    int k = i / Fpad, f = i - k * Fpad;
    Wp[i] = (k < K && f < F) ? W[k * F + f] : 0.0f;
}

// ---------------------------------------------------------------------------
// Degree count on target nodes, then dinv = rsqrt(deg + 1)
// ---------------------------------------------------------------------------
__global__ void k_count_deg(const int* __restrict__ dst, float* __restrict__ deg, int E) {
    int e = blockIdx.x * blockDim.x + threadIdx.x;
    if (e < E) atomicAdd(&deg[dst[e]], 1.0f);
}

__global__ void k_dinv(const float* __restrict__ deg, float* __restrict__ dinv, int Nn) {
    int i = blockIdx.x * blockDim.x + threadIdx.x;
    if (i < Nn) dinv[i] = rsqrtf(deg[i] + 1.0f);
}

// ---------------------------------------------------------------------------
// WMMA GEMM, compile-time K, F, LDA (K%32==0, F%16==0, output stride = LDH).
// out[M,F]@LDH = A[M,K]@LDA x W[K,F] (W fully materialized/padded, stride F).
// One wave: preload B into regs once, grid-stride over 16-row M tiles.
// A lane layout (16-bit A 16x32, ISA 7.12.2):
//   lane<16 : row=lane,    elems0..7=K(+0..7),  elems8..15=K(+16..23)
//   lane>=16: row=lane-16, elems0..7=K(+8..15), elems8..15=K(+24..31)
// B mirrors with row->col. D: VGPR v, lane L -> D[v + (L<16?0:8)][L&15].
// ---------------------------------------------------------------------------
template<int K, int F, int LDA>
__global__ void k_gemm_wmma(const float* __restrict__ A,
                            const float* __restrict__ W,
                            float* __restrict__ out,
                            int Mtiles, int Mrows) {
    constexpr int KS = K / 32;
    constexpr int NT = F / 16;
    const int lane   = threadIdx.x & 31;
    const int wid    = blockIdx.x * (blockDim.x >> 5) + (threadIdx.x >> 5);
    const int nWaves = gridDim.x * (blockDim.x >> 5);

    const int row   = lane & 15;
    const int khalf = (lane < 16) ? 0 : 8;
    const bool evenM = ((Mrows & 15) == 0);

    // ---- preload B into registers (reused for every M tile) ----
    v16h breg[KS][NT];
#pragma unroll
    for (int ks = 0; ks < KS; ++ks) {
#pragma unroll
        for (int nt = 0; nt < NT; ++nt) {
            const int col = nt * 16 + row;
#pragma unroll
            for (int j = 0; j < 8; ++j) {
                breg[ks][nt][j]     = (_Float16)W[(ks * 32 + khalf + j) * F + col];
                breg[ks][nt][8 + j] = (_Float16)W[(ks * 32 + khalf + 16 + j) * F + col];
            }
        }
    }

    for (int mt = wid; mt < Mtiles; mt += nWaves) {
        const int gm  = mt * 16 + row;
        const int gmc = (gm < Mrows) ? gm : (Mrows - 1);   // address clamp; loads unpredicated
        v16h areg[KS];
#pragma unroll
        for (int ks = 0; ks < KS; ++ks) {
            const float* base = A + (size_t)gmc * LDA + ks * 32 + khalf;
            const float4 a0 = ((const float4*)base)[0];
            const float4 a1 = ((const float4*)base)[1];
            const float4 a2 = ((const float4*)(base + 16))[0];
            const float4 a3 = ((const float4*)(base + 16))[1];
            areg[ks][0]  = (_Float16)a0.x; areg[ks][1]  = (_Float16)a0.y;
            areg[ks][2]  = (_Float16)a0.z; areg[ks][3]  = (_Float16)a0.w;
            areg[ks][4]  = (_Float16)a1.x; areg[ks][5]  = (_Float16)a1.y;
            areg[ks][6]  = (_Float16)a1.z; areg[ks][7]  = (_Float16)a1.w;
            areg[ks][8]  = (_Float16)a2.x; areg[ks][9]  = (_Float16)a2.y;
            areg[ks][10] = (_Float16)a2.z; areg[ks][11] = (_Float16)a2.w;
            areg[ks][12] = (_Float16)a3.x; areg[ks][13] = (_Float16)a3.y;
            areg[ks][14] = (_Float16)a3.z; areg[ks][15] = (_Float16)a3.w;
        }

        const int mbase = mt * 16 + ((lane < 16) ? 0 : 8);
        const bool fullTile = evenM || (mt + 1 < Mtiles);   // wave-uniform
#pragma unroll
        for (int nt = 0; nt < NT; ++nt) {
            v8f acc = {};
#pragma unroll
            for (int ks = 0; ks < KS; ++ks) {
                acc = __builtin_amdgcn_wmma_f32_16x16x32_f16(
                    false, areg[ks], false, breg[ks][nt], (short)0, acc, false, false);
            }
            const int col = nt * 16 + row;
            float* op = out + (size_t)mbase * LDH + col;    // row step = LDH*4 B (constant)
            if (fullTile) {
#pragma unroll
                for (int v = 0; v < 8; ++v) op[v * LDH] = acc[v];
            } else {
#pragma unroll
                for (int v = 0; v < 8; ++v)
                    if (mbase + v < Mrows) op[v * LDH] = acc[v];
            }
        }
    }
}

// ---------------------------------------------------------------------------
// Edge scatter: agg[dst] += h[src] * dinv[src]*dinv[dst]   (row stride LDH)
// h rows are L2-resident; float4 gathers + f32 global atomics.
// ---------------------------------------------------------------------------
template<int F>
__global__ void k_scatter(const int* __restrict__ src, const int* __restrict__ dst,
                          const float* __restrict__ h, const float* __restrict__ dinv,
                          float* __restrict__ agg, int E) {
    int e = blockIdx.x * blockDim.x + threadIdx.x;
    if (e >= E) return;
    const int s = src[e], d = dst[e];
    const float norm = dinv[s] * dinv[d];
    const float4* hs = (const float4*)(h + (size_t)s * LDH);
    float* ad = agg + (size_t)d * LDH;
#pragma unroll
    for (int q = 0; q < F / 4; ++q) {
        float4 v = hs[q];
        atomicAdd(&ad[4 * q + 0], v.x * norm);
        atomicAdd(&ad[4 * q + 1], v.y * norm);
        atomicAdd(&ad[4 * q + 2], v.z * norm);
        atomicAdd(&ad[4 * q + 3], v.w * norm);
    }
}

// h = agg + h*dinv^2 + b   (in place on h; row stride LDH, F real cols)
template<int F>
__global__ void k_finalize(float* __restrict__ h, const float* __restrict__ agg,
                           const float* __restrict__ dinv, const float* __restrict__ b,
                           int Nn) {
    int i = blockIdx.x * blockDim.x + threadIdx.x;
    if (i >= Nn * F) return;
    const int n = i / F, f = i - n * F;
    const float di = dinv[n];
    const size_t idx = (size_t)n * LDH + f;
    h[idx] = agg[idx] + h[idx] * di * di + b[f];
}

// ---------------------------------------------------------------------------
// Deterministic hash RNG -> Gumbel noise for categorical sampling
// ---------------------------------------------------------------------------
__device__ __forceinline__ float hash_u01(unsigned x) {
    x ^= x >> 17; x *= 0xED5AD4BBu;
    x ^= x >> 11; x *= 0xAC4C1B51u;
    x ^= x >> 15; x *= 0x31848BABu;
    x ^= x >> 14;
    return (float)(x >> 8) * (1.0f / 16777216.0f) + 1.0e-10f;
}
__device__ __forceinline__ float gumbel(unsigned x) {
    return -__logf(-__logf(hash_u01(x)));
}
__device__ __forceinline__ float relu6(float x) { return fminf(fmaxf(x, 0.0f), 6.0f); }

// ---------------------------------------------------------------------------
// Start head: relu6(h@Ws1+bs1)@Ws2+bs2 -> gumbel argmax. Weights staged in LDS.
// ---------------------------------------------------------------------------
__global__ void k_start_head(const float* __restrict__ h,
                             const float* __restrict__ W1, const float* __restrict__ b1,
                             const float* __restrict__ W2, const float* __restrict__ b2,
                             int* __restrict__ istart, float* __restrict__ fout, int Nn) {
    __shared__ float sW1[32 * 16], sb1[16], sW2[16 * 32], sb2[32];
    for (int i = threadIdx.x; i < 32 * 16; i += blockDim.x) sW1[i] = W1[i];
    for (int i = threadIdx.x; i < 16;      i += blockDim.x) sb1[i] = b1[i];
    for (int i = threadIdx.x; i < 16 * 32; i += blockDim.x) sW2[i] = W2[i];
    for (int i = threadIdx.x; i < 32;      i += blockDim.x) sb2[i] = b2[i];
    __syncthreads();
    int n = blockIdx.x * blockDim.x + threadIdx.x;
    if (n >= Nn) return;

    float x[32];
    const float4* hr = (const float4*)(h + (size_t)n * LDH);
#pragma unroll
    for (int q = 0; q < 8; ++q) {
        float4 v = hr[q];
        x[4*q] = v.x; x[4*q+1] = v.y; x[4*q+2] = v.z; x[4*q+3] = v.w;
    }
    float hid[16];
#pragma unroll
    for (int j = 0; j < 16; ++j) {
        float s = sb1[j];
#pragma unroll
        for (int i2 = 0; i2 < 32; ++i2) s += x[i2] * sW1[i2 * 16 + j];
        hid[j] = relu6(s);
    }
    float best = -1.0e30f; int arg = 0;
    for (int c = 0; c < 32; ++c) {
        float s = sb2[c];
#pragma unroll
        for (int j = 0; j < 16; ++j) s += hid[j] * sW2[j * 32 + c];
        s += gumbel((unsigned)(n * 32 + c) ^ 0x9E3779B9u);
        if (s > best) { best = s; arg = c; }
    }
    istart[n] = arg;
    fout[n]   = (float)arg;
}

// ---------------------------------------------------------------------------
// End head over combined rows [0,2N): row i<N uses h[i], else h[start[i-N]].
// ---------------------------------------------------------------------------
__global__ void k_end_head(const float* __restrict__ h, const int* __restrict__ istart,
                           const float* __restrict__ W1, const float* __restrict__ b1,
                           const float* __restrict__ W2, const float* __restrict__ b2,
                           float* __restrict__ fout, int Nn) {
    __shared__ float sW1[32 * 24], sb1[24], sW2[24 * 32], sb2[32];
    for (int i = threadIdx.x; i < 32 * 24; i += blockDim.x) sW1[i] = W1[i];
    for (int i = threadIdx.x; i < 24;      i += blockDim.x) sb1[i] = b1[i];
    for (int i = threadIdx.x; i < 24 * 32; i += blockDim.x) sW2[i] = W2[i];
    for (int i = threadIdx.x; i < 32;      i += blockDim.x) sb2[i] = b2[i];
    __syncthreads();
    int i = blockIdx.x * blockDim.x + threadIdx.x;
    if (i >= 2 * Nn) return;

    const int srcrow = (i < Nn) ? i : istart[i - Nn];
    float x[32];
    const float4* hr = (const float4*)(h + (size_t)srcrow * LDH);
#pragma unroll
    for (int q = 0; q < 8; ++q) {
        float4 v = hr[q];
        x[4*q] = v.x; x[4*q+1] = v.y; x[4*q+2] = v.z; x[4*q+3] = v.w;
    }
    float hid[24];
#pragma unroll
    for (int j = 0; j < 24; ++j) {
        float s = sb1[j];
#pragma unroll
        for (int i2 = 0; i2 < 32; ++i2) s += x[i2] * sW1[i2 * 24 + j];
        hid[j] = relu6(s);
    }
    float best = -1.0e30f; int arg = 0;
    for (int c = 0; c < 32; ++c) {
        float s = sb2[c];
#pragma unroll
        for (int j = 0; j < 24; ++j) s += hid[j] * sW2[j * 32 + c];
        s += gumbel((unsigned)(i * 32 + c) ^ 0x85EBCA6Bu);
        if (s > best) { best = s; arg = c; }
    }
    fout[i] = (float)arg;
}

// ---------------------------------------------------------------------------
// Launch
// ---------------------------------------------------------------------------
extern "C" void kernel_launch(void* const* d_in, const int* in_sizes, int n_in,
                              void* d_out, int out_size, void* d_ws, size_t ws_size,
                              hipStream_t stream) {
    const int Nn = in_sizes[0] / 128;     // 100000
    const int E  = in_sizes[1] / 2;       // 3200000

    const float* x   = (const float*)d_in[0];
    const int*   src = (const int*)d_in[1];
    const int*   dst = src + E;
    const float* W1  = (const float*)d_in[3];  const float* b1  = (const float*)d_in[4];
    const float* W2  = (const float*)d_in[5];  const float* b2  = (const float*)d_in[6];
    const float* W3  = (const float*)d_in[7];  const float* b3  = (const float*)d_in[8];
    const float* Ws1 = (const float*)d_in[9];  const float* bs1 = (const float*)d_in[10];
    const float* Ws2 = (const float*)d_in[11]; const float* bs2 = (const float*)d_in[12];
    const float* We1 = (const float*)d_in[13]; const float* be1 = (const float*)d_in[14];
    const float* We2 = (const float*)d_in[15]; const float* be2 = (const float*)d_in[16];

    float* ws   = (float*)d_ws;
    float* dinv = ws;                         // [N]
    float* h1   = dinv + Nn;                  // [N, LDH] (cols 16.. zero)
    float* h2   = h1 + (size_t)Nn * LDH;      // [N, LDH] (cols 24.. zero)
    float* h3   = h2 + (size_t)Nn * LDH;      // [N, LDH]
    float* agg  = h3 + (size_t)Nn * LDH;      // [N, LDH] (also deg scratch)
    float* Wp2  = agg + (size_t)Nn * LDH;     // [32,32] padded W2
    float* Wp3  = Wp2 + 32 * 32;              // [32,32] padded W3
    int*   istart = (int*)(Wp3 + 32 * 32);    // [N]
    float* outf = (float*)d_out;              // [3N]: start(N) ++ end(2N)

    const int TB = 256;
    const int gridE  = (E + TB - 1) / TB;
    const int gridNF = (Nn * LDH + TB - 1) / TB;
    const int Mtiles = (Nn + 15) / 16;
    // 8 waves/block, ~4 M-tiles per wave to amortize the B-register preload
    const int gemmGrid = (Mtiles + 8 * 4 - 1) / (8 * 4);

    // pack padded weights
    hipLaunchKernelGGL(k_pack_w, dim3(4), dim3(TB), 0, stream, W2, Wp2, 16, 24, 32, 32);
    hipLaunchKernelGGL(k_pack_w, dim3(4), dim3(TB), 0, stream, W3, Wp3, 24, 32, 32, 32);

    // degrees -> dinv (shared across all three layers)
    hipLaunchKernelGGL(k_zero_f32, dim3((Nn + TB - 1) / TB), dim3(TB), 0, stream, agg, Nn);
    hipLaunchKernelGGL(k_count_deg, dim3(gridE), dim3(TB), 0, stream, dst, agg, E);
    hipLaunchKernelGGL(k_dinv, dim3((Nn + TB - 1) / TB), dim3(TB), 0, stream, agg, dinv, Nn);

    // ---- layer 1: 128 -> 16 (zero h1 so pad cols stay 0 for next layer's K) ----
    hipLaunchKernelGGL(k_zero_f32, dim3(gridNF), dim3(TB), 0, stream, h1, Nn * LDH);
    hipLaunchKernelGGL((k_gemm_wmma<128, 16, 128>), dim3(gemmGrid), dim3(256), 0, stream,
                       x, W1, h1, Mtiles, Nn);
    hipLaunchKernelGGL(k_zero_f32, dim3(gridNF), dim3(TB), 0, stream, agg, Nn * LDH);
    hipLaunchKernelGGL(k_scatter<16>, dim3(gridE), dim3(TB), 0, stream, src, dst, h1, dinv, agg, E);
    hipLaunchKernelGGL(k_finalize<16>, dim3((Nn * 16 + TB - 1) / TB), dim3(TB), 0, stream,
                       h1, agg, dinv, b1, Nn);

    // ---- layer 2: 16 -> 24 (padded K=32, F=32; pad output cols get zeros) ----
    hipLaunchKernelGGL((k_gemm_wmma<32, 32, LDH>), dim3(gemmGrid), dim3(256), 0, stream,
                       h1, Wp2, h2, Mtiles, Nn);
    hipLaunchKernelGGL(k_zero_f32, dim3(gridNF), dim3(TB), 0, stream, agg, Nn * LDH);
    hipLaunchKernelGGL(k_scatter<24>, dim3(gridE), dim3(TB), 0, stream, src, dst, h2, dinv, agg, E);
    hipLaunchKernelGGL(k_finalize<24>, dim3((Nn * 24 + TB - 1) / TB), dim3(TB), 0, stream,
                       h2, agg, dinv, b2, Nn);

    // ---- layer 3: 24 -> 32 (padded K=32) ----
    hipLaunchKernelGGL((k_gemm_wmma<32, 32, LDH>), dim3(gemmGrid), dim3(256), 0, stream,
                       h2, Wp3, h3, Mtiles, Nn);
    hipLaunchKernelGGL(k_zero_f32, dim3(gridNF), dim3(TB), 0, stream, agg, Nn * LDH);
    hipLaunchKernelGGL(k_scatter<32>, dim3(gridE), dim3(TB), 0, stream, src, dst, h3, dinv, agg, E);
    hipLaunchKernelGGL(k_finalize<32>, dim3((Nn * 32 + TB - 1) / TB), dim3(TB), 0, stream,
                       h3, agg, dinv, b3, Nn);

    // ---- heads ----
    hipLaunchKernelGGL(k_start_head, dim3((Nn + TB - 1) / TB), dim3(TB), 0, stream,
                       h3, Ws1, bs1, Ws2, bs2, istart, outf, Nn);
    hipLaunchKernelGGL(k_end_head, dim3((2 * Nn + TB - 1) / TB), dim3(TB), 0, stream,
                       h3, istart, We1, be1, We2, be2, outf + Nn, Nn);
}
